// SegSmall_23914377904592
// MI455X (gfx1250) — matrix operand order repaired and checked by reading, not compiled, not run on hardware
//
#include <hip/hip_runtime.h>
#include <hip/hip_bf16.h>

// ---------------------------------------------------------------------------
// PtConv (SegSmall) fused kernel for MI455X / gfx1250 (wave32, WMMA f16)
//   B=4, N=16384, K=16 neighbors, C_IN=C_OUT=64, KS=16, DIM=3
// All matmuls use v_wmma_f32_16x16x32_f16 (f32 accumulate).
// Weight [1024,64] staged once per workgroup into LDS, pre-swizzled into the
// WMMA B-operand lane layout, so the dominant K=1024 GEMM streams LDS->WMMA.
// ---------------------------------------------------------------------------

typedef __attribute__((ext_vector_type(16))) _Float16 v16h;
typedef __attribute__((ext_vector_type(8)))  _Float16 v8h;
typedef __attribute__((ext_vector_type(8)))  float    v8f;

#define WMMA16(A, B, C) \
  __builtin_amdgcn_wmma_f32_16x16x32_f16(false, (A), false, (B), (short)0, (C), false, false)

__device__ __forceinline__ v16h load_a16(const _Float16* p0, const _Float16* p1) {
  // A-operand: elements 0..7 = first K-group, 8..15 = second K-group
  v8h lo = *(const v8h*)p0;
  v8h hi = *(const v8h*)p1;
  v16h r;
#pragma unroll
  for (int i = 0; i < 8; ++i) { r[i] = lo[i]; r[i + 8] = hi[i]; }
  return r;
}

__device__ __forceinline__ v16h load_b16(const _Float16* p) {
  return *(const v16h*)p;  // 32B contiguous, pre-swizzled B layout
}

__device__ __forceinline__ v8f bcast8(float x) {
  v8f r;
#pragma unroll
  for (int i = 0; i < 8; ++i) r[i] = x;
  return r;
}

constexpr int Bc   = 4;
constexpr int Nc   = 16384;
constexpr int NW   = 4;                  // waves per workgroup
constexpr int NTILES = (Bc * Nc) / 16;   // 4096 tiles of 16 points

__global__ __launch_bounds__(128, 1)
void ptconv_wmma_kernel(const float* __restrict__ features,
                        const float* __restrict__ input_pts,
                        const float* __restrict__ output_pts,
                        const float* __restrict__ weight,
                        const float* __restrict__ bias,
                        const float* __restrict__ centers,
                        const float* __restrict__ w1, const float* __restrict__ b1,
                        const float* __restrict__ w2, const float* __restrict__ b2,
                        const float* __restrict__ w3, const float* __restrict__ b3,
                        const int*   __restrict__ indices,
                        float* __restrict__ out) {
  // ---- LDS ----------------------------------------------------------------
  // Main weight, pre-swizzled B tiles: [kk(32) * nt(4)][lane(32)][t(16)]
  // K-order is the agg layout order: K' = j*64 + c  ->  weight[(c*16+j)*64+o]
  __shared__ __attribute__((aligned(32))) _Float16 sW[128 * 512];       // 128 KB
  // MLP weight B tiles: 4 (w1: ks*2+nt) + 1 (w2) + 1 (w3, K-padded)
  __shared__ __attribute__((aligned(32))) _Float16 sMLPw[6 * 512];      // 6 KB
  // Per-wave scratch
  __shared__ __attribute__((aligned(32))) _Float16 sAgg[NW][16 * 1024]; // 128 KB
  __shared__ __attribute__((aligned(32))) _Float16 sD48[NW][16 * 64];   // 8 KB
  __shared__ __attribute__((aligned(32))) _Float16 sH  [NW][16 * 32];   // 4 KB
  __shared__ __attribute__((aligned(32))) _Float16 sDT [NW][16 * 16];   // 2 KB
  __shared__ __attribute__((aligned(32))) _Float16 sFT [NW][64 * 16];   // 8 KB
  __shared__ __attribute__((aligned(32))) _Float16 sZero[16];           // 32 B
  __shared__ float sB1[32], sB2[16], sB3[16], sCent[48], sBias[64];

  // ---- One-time staging (cooperative, 128 threads) ------------------------
  // B-operand swizzle: lane<16 holds col = nt*16+lane,   K = kk*32 + (t&7) + 16*(t>>3)
  //                    lane>=16 holds col = nt*16+ln-16, K = kk*32 + 8 + (t&7) + 16*(t>>3)
  for (int i = threadIdx.x; i < 128 * 512; i += blockDim.x) {
    int tile = i >> 9;
    int t16  = i & 511;
    int ln   = t16 >> 4;
    int t    = t16 & 15;
    int kk = tile >> 2, nt = tile & 3;
    int col  = nt * 16 + (ln & 15);
    int kl   = (t & 7) + ((ln < 16) ? 0 : 8) + ((t >> 3) << 4);
    int kpr  = kk * 32 + kl;          // K' in agg order: j*64 + c
    int c    = kpr & 63;
    int j    = kpr >> 6;
    sW[i] = (_Float16)weight[(c * 16 + j) * 64 + col];
  }
  for (int i = threadIdx.x; i < 6 * 512; i += blockDim.x) {
    int tile = i >> 9;
    int t16  = i & 511;
    int ln   = t16 >> 4;
    int t    = t16 & 15;
    int kl   = (t & 7) + ((ln < 16) ? 0 : 8) + ((t >> 3) << 4);
    int c16  = ln & 15;
    float v = 0.f;
    if (tile < 4) {                 // w1 [48,32], K padded to 64
      int ks = tile >> 1, nt = tile & 1;
      int kr = ks * 32 + kl;
      if (kr < 48) v = w1[kr * 32 + nt * 16 + c16];
    } else if (tile == 4) {         // w2 [32,16]
      v = w2[kl * 16 + c16];
    } else {                        // w3 [16,16], K padded to 32
      if (kl < 16) v = w3[kl * 16 + c16];
    }
    sMLPw[i] = (_Float16)v;
  }
  if (threadIdx.x < 64) sBias[threadIdx.x] = bias[threadIdx.x];
  if (threadIdx.x < 48) sCent[threadIdx.x] = centers[threadIdx.x];
  if (threadIdx.x < 32) sB1[threadIdx.x] = b1[threadIdx.x];
  if (threadIdx.x < 16) {
    sB2[threadIdx.x] = b2[threadIdx.x];
    sB3[threadIdx.x] = b3[threadIdx.x];
    sZero[threadIdx.x] = (_Float16)0.f;
  }
  __syncthreads();

  // ---- Per-wave main loop -------------------------------------------------
  const int w    = threadIdx.x >> 5;
  const int lane = threadIdx.x & 31;
  const int r16  = lane & 15;
  const int hiw  = lane >> 4;          // 0 for lanes 0-15, 1 for 16-31

  _Float16* agg  = sAgg[w];
  _Float16* d48  = sD48[w];
  _Float16* hbuf = sH[w];
  _Float16* dT   = sDT[w];
  _Float16* fT   = sFT[w];

  for (int tile = blockIdx.x * NW + w; tile < NTILES; tile += gridDim.x * NW) {
    const int p0 = tile * 16;          // global point index (flattened B*N)
    const int bb = p0 / Nc;            // batch

    for (int m = 0; m < 16; ++m) {
      const int pg = p0 + m;
      // prefetch next point's gather metadata (global_prefetch_b8)
      __builtin_prefetch(indices + (pg + 1) * 16, 0, 0);
      __builtin_prefetch(output_pts + (pg + 1) * 3, 0, 0);

      const float ox = output_pts[pg * 3 + 0];
      const float oy = output_pts[pg * 3 + 1];
      const float oz = output_pts[pg * 3 + 2];

      // Build MLP input rows D48[k][64] (cols 48..63 zero): lanes 0..15
      if (lane < 16) {
        const int k   = lane;
        const int idx = indices[pg * 16 + k];
        const float* ip = input_pts + ((size_t)bb * Nc + idx) * 3;
        const float px = ip[0] - ox, py = ip[1] - oy, pz = ip[2] - oz;
        _Float16* row = d48 + k * 64;
        v8h cx0, cx1, cy0, cy1, cz0, cz1, z8 = {};
#pragma unroll
        for (int i = 0; i < 8; ++i) {
          cx0[i] = (_Float16)(px - sCent[i]);
          cx1[i] = (_Float16)(px - sCent[8 + i]);
          cy0[i] = (_Float16)(py - sCent[16 + i]);
          cy1[i] = (_Float16)(py - sCent[24 + i]);
          cz0[i] = (_Float16)(pz - sCent[32 + i]);
          cz1[i] = (_Float16)(pz - sCent[40 + i]);
        }
        *(v8h*)(row + 0)  = cx0;  *(v8h*)(row + 8)  = cx1;
        *(v8h*)(row + 16) = cy0;  *(v8h*)(row + 24) = cy1;
        *(v8h*)(row + 32) = cz0;  *(v8h*)(row + 40) = cz1;
        *(v8h*)(row + 48) = z8;   *(v8h*)(row + 56) = z8;
      }
      // Gather neighbor features -> fT[c*16+k] (transposed, K contiguous)
      for (int k = 0; k < 16; ++k) {
        const int idx = indices[pg * 16 + k];
        const float* fp = features + ((size_t)bb * Nc + idx) * 64 + lane * 2;
        const float f0 = fp[0], f1 = fp[1];
        fT[(lane * 2) * 16 + k]     = (_Float16)f0;
        fT[(lane * 2 + 1) * 16 + k] = (_Float16)f1;
      }

      // --- MLP layer 1: [16,64pad] @ [64pad,32]  (bias in C operand) ---
      {
        const _Float16* rp = d48 + r16 * 64 + hiw * 8;
        v16h a0 = load_a16(rp, rp + 16);
        v16h a1 = load_a16(rp + 32, rp + 48);
#pragma unroll
        for (int nt = 0; nt < 2; ++nt) {
          v8f c = bcast8(sB1[nt * 16 + r16]);
          c = WMMA16(a0, load_b16(sMLPw + (0 * 2 + nt) * 512 + lane * 16), c);
          c = WMMA16(a1, load_b16(sMLPw + (1 * 2 + nt) * 512 + lane * 16), c);
#pragma unroll
          for (int r = 0; r < 8; ++r) {
            hbuf[(hiw * 8 + r) * 32 + nt * 16 + r16] = (_Float16)fmaxf(c[r], 0.f);
          }
        }
      }
      // --- MLP layer 2: [16,32] @ [32,16] ---
      {
        const _Float16* rp = hbuf + r16 * 32 + hiw * 8;
        v16h a = load_a16(rp, rp + 16);
        v8f c = WMMA16(a, load_b16(sMLPw + 4 * 512 + lane * 16), bcast8(sB2[r16]));
#pragma unroll
        for (int r = 0; r < 8; ++r) {
          hbuf[(hiw * 8 + r) * 32 + r16]      = (_Float16)fmaxf(c[r], 0.f);
          hbuf[(hiw * 8 + r) * 32 + 16 + r16] = (_Float16)0.f;  // K pad
        }
      }
      // --- MLP layer 3: [16,32pad] @ [32pad,16] -> dT[j][k] (one b128) ---
      {
        const _Float16* rp = hbuf + r16 * 32 + hiw * 8;
        v16h a = load_a16(rp, rp + 16);
        v8f c = WMMA16(a, load_b16(sMLPw + 5 * 512 + lane * 16), bcast8(sB3[r16]));
        v8h d8;
#pragma unroll
        for (int r = 0; r < 8; ++r) d8[r] = (_Float16)fmaxf(c[r], 0.f);
        *(v8h*)(dT + r16 * 16 + hiw * 8) = d8;   // D: M=k=hiw*8+r, N=j=r16
      }
      // --- Step 5: agg[64,16] = feat^T[64,16] @ d[16,16] (K padded to 32) ---
      //     stored as agg[m][j*64 + c] so each D octet is one b128 store
      {
        const _Float16* bp = dT + r16 * 16 + hiw * 8;
        v16h b5 = load_a16(bp, sZero);   // elements 8..15 hold K=16..31 == 0
#pragma unroll
        for (int ct = 0; ct < 4; ++ct) {
          v16h a = load_a16(fT + (ct * 16 + r16) * 16 + hiw * 8, sZero);
          v8f c = WMMA16(a, b5, bcast8(0.f));
          v8h a8;
#pragma unroll
          for (int r = 0; r < 8; ++r) a8[r] = (_Float16)c[r];
          // c index = ct*16 + hiw*8 + r (contiguous in r), j = r16
          *(v8h*)(agg + m * 1024 + r16 * 64 + ct * 16 + hiw * 8) = a8;
        }
      }
    }  // point loop

    // --- Step 6: OUT[16,64] = AGG[16,1024] @ W'[1024,64], 32 k-steps ---
    //     bias*K folded into C (epilogue multiplies by 1/K)
    v8f acc[4];
#pragma unroll
    for (int nt = 0; nt < 4; ++nt) acc[nt] = bcast8(sBias[nt * 16 + r16] * 16.f);
#pragma unroll 2
    for (int kk = 0; kk < 32; ++kk) {
      const _Float16* rp = agg + r16 * 1024 + kk * 32 + hiw * 8;
      v16h a = load_a16(rp, rp + 16);
#pragma unroll
      for (int nt = 0; nt < 4; ++nt) {
        acc[nt] = WMMA16(a, load_b16(sW + (kk * 4 + nt) * 512 + lane * 16), acc[nt]);
      }
    }
    // Epilogue: /K (=16), scatter to global (each store: 2x64B segments)
#pragma unroll
    for (int nt = 0; nt < 4; ++nt) {
      const int o = nt * 16 + r16;
#pragma unroll
      for (int r = 0; r < 8; ++r) {
        const int m = hiw * 8 + r;
        out[(size_t)(p0 + m) * 64 + o] = acc[nt][r] * 0.0625f;
      }
    }
  }  // tile loop
}

extern "C" void kernel_launch(void* const* d_in, const int* in_sizes, int n_in,
                              void* d_out, int out_size, void* d_ws, size_t ws_size,
                              hipStream_t stream) {
  (void)in_sizes; (void)n_in; (void)out_size; (void)d_ws; (void)ws_size;
  const float* features   = (const float*)d_in[0];
  const float* input_pts  = (const float*)d_in[1];
  const float* output_pts = (const float*)d_in[2];
  const float* weight     = (const float*)d_in[3];
  const float* bias       = (const float*)d_in[4];
  const float* centers    = (const float*)d_in[5];
  const float* w1         = (const float*)d_in[6];
  const float* b1         = (const float*)d_in[7];
  const float* w2         = (const float*)d_in[8];
  const float* b2         = (const float*)d_in[9];
  const float* w3         = (const float*)d_in[10];
  const float* b3         = (const float*)d_in[11];
  const int*   indices    = (const int*)d_in[12];
  float* outp = (float*)d_out;

  dim3 grid(256), block(128);  // 256 WGs * 4 waves -> 4 tiles per wave
  ptconv_wmma_kernel<<<grid, block, 0, stream>>>(
      features, input_pts, output_pts, weight, bias, centers,
      w1, b1, w2, b2, w3, b3, indices, outp);
}
